// GATEncoder_51281909514734
// MI455X (gfx1250) — matrix operand order repaired and checked by reading, not compiled, not run on hardware
//
#include <hip/hip_runtime.h>

typedef __attribute__((ext_vector_type(16))) _Float16 v16h;
typedef __attribute__((ext_vector_type(8)))  _Float16 v8h;
typedef __attribute__((ext_vector_type(8)))  float    v8f;

#define N_NODES 50000
#define N_EDGES 800000
#define HID     128
#define NHEAD   4
#define CHEAD   32
#define NODE_IN 22
#define EDGE_IN 12
#define NGRAPH  256
#define EA      (N_EDGES + N_NODES)   // 850000, divisible by 16
#define SLOPE   0.2f

// ---------------------------------------------------------------------------
// WMMA fragment helpers (wave32; A/B use contiguous k order, not the ISA table order)
// A: 16x32 f16 (MxK). lane l: m=l&15, h=l>>4; a[j]=A[m][k0+h*16+j]
//   -> one contiguous 32B chunk per lane: 2x ds_load_b128
// B: 32x16 f16 (KxN). lane l: n=l&15, h=l>>4; b[j]=B[h*16+j][n]
//   -> pre-swizzled in LDS so each lane reads 32 contiguous bytes
// C/D: 16x16 f32.     lane l: n=l&15, h=l>>4; c[r]=C[h*8+r][n]
// ---------------------------------------------------------------------------
__device__ __forceinline__ v16h load_a_frag(const _Float16* sm, int ld, int k0) {
  int l = threadIdx.x & 31;
  int m = l & 15, hh = l >> 4;
  const _Float16* p = sm + m * ld + k0 + hh * 16;
  v16h a;
  ((v8h*)&a)[0] = *(const v8h*)(p);
  ((v8h*)&a)[1] = *(const v8h*)(p + 8);
  return a;
}

// slot of element B[k][n] (k < kpad, n < 128) inside swizzled buffer:
// fragment (k0=k&~31, n0=n&~15), lane = ((k>>4)&1)*16 + (n&15), j = k&15
__device__ __forceinline__ int bswz(int k, int n) {
  int frag = ((k >> 5) << 3) + (n >> 4);              // ktile*8 + ntile
  int lane = (((k >> 4) & 1) << 4) + (n & 15);
  return (frag << 9) + (lane << 4) + (k & 15);
}

__device__ __forceinline__ v16h load_b_swz(const _Float16* swz, int k0, int n0) {
  int l = threadIdx.x & 31;
  const _Float16* p = swz + ((((k0 >> 5) << 3) + (n0 >> 4)) << 9) + (l << 4);
  v16h b;
  ((v8h*)&b)[0] = *(const v8h*)(p);
  ((v8h*)&b)[1] = *(const v8h*)(p + 8);
  return b;
}

__device__ __forceinline__ v8f wmma_f16(v16h a, v16h b, v8f c) {
  return __builtin_amdgcn_wmma_f32_16x16x32_f16(false, a, false, b, (short)0, c,
                                                false, false);
}

// Monotonic float atomic max via int/uint trick
__device__ __forceinline__ void atomicMaxF(float* addr, float v) {
  int vi = __float_as_int(v);
  if (vi >= 0) atomicMax((int*)addr, vi);
  else         atomicMin((unsigned int*)addr, (unsigned int)vi);
}

__device__ __forceinline__ float siluf(float x) { return x / (1.f + __expf(-x)); }

// ---------------------------------------------------------------------------
__global__ void k_fill(float* __restrict__ p, float v, size_t n) {
  for (size_t i = (size_t)blockIdx.x * blockDim.x + threadIdx.x; i < n;
       i += (size_t)gridDim.x * blockDim.x)
    p[i] = v;
}

// e = silu(edge_attr @ Wp + bp); write e into e_a[0:E); accumulate scatter-mean
__global__ void k_edgeproj(const float* __restrict__ eattr, const float* __restrict__ Wp,
                           const float* __restrict__ bp, const int* __restrict__ ei,
                           _Float16* __restrict__ ea, float* __restrict__ nsum,
                           float* __restrict__ cnt) {
  __shared__ float Wsm[EDGE_IN * HID];
  __shared__ float bsm[HID];
  __shared__ float attr_sm[64 * EDGE_IN];
  __shared__ int   dsm[64];
  int tid = threadIdx.x;
  for (int i = tid; i < EDGE_IN * HID; i += blockDim.x) Wsm[i] = Wp[i];
  if (tid < HID) bsm[tid] = bp[tid];
  int base = blockIdx.x * 64;
  for (int i = tid; i < 64 * EDGE_IN; i += blockDim.x) {
    int e = base + i / EDGE_IN;
    attr_sm[i] = (e < N_EDGES) ? eattr[(size_t)e * EDGE_IN + (i % EDGE_IN)] : 0.f;
  }
  for (int i = tid; i < 64; i += blockDim.x) {
    int e = base + i;
    dsm[i] = (e < N_EDGES) ? ei[N_EDGES + e] : 0;
  }
  __syncthreads();
  for (int i = tid; i < 64 * HID; i += blockDim.x) {
    int el = i >> 7, c = i & 127;
    int e = base + el;
    if (e < N_EDGES) {
      float s = bsm[c];
#pragma unroll
      for (int k = 0; k < EDGE_IN; ++k) s += attr_sm[el * EDGE_IN + k] * Wsm[k * HID + c];
      float v = siluf(s);
      ea[(size_t)e * HID + c] = (_Float16)v;
      int d = dsm[el];
      atomicAdd(&nsum[(size_t)d * HID + c], v);
      if (c == 0) atomicAdd(&cnt[d], 1.f);
    }
  }
}

// self-loop edge attr = scatter-mean; write into e_a[E:E+N)
__global__ void k_selfloop(const float* __restrict__ nsum, const float* __restrict__ cnt,
                           _Float16* __restrict__ ea) {
  size_t total = (size_t)N_NODES * HID;
  for (size_t i = (size_t)blockIdx.x * blockDim.x + threadIdx.x; i < total;
       i += (size_t)gridDim.x * blockDim.x) {
    int n = (int)(i >> 7), c = (int)(i & 127);
    float v = nsum[i] / fmaxf(cnt[n], 1.f);
    ea[(size_t)(N_EDGES + n) * HID + c] = (_Float16)v;
  }
}

// pad x [N,22] -> f16 [N,32]
__global__ void k_xpad(const float* __restrict__ x, _Float16* __restrict__ xp) {
  size_t total = (size_t)N_NODES * 32;
  for (size_t i = (size_t)blockIdx.x * blockDim.x + threadIdx.x; i < total;
       i += (size_t)gridDim.x * blockDim.x) {
    int n = (int)(i >> 5), k = (int)(i & 31);
    xp[i] = (k < NODE_IN) ? (_Float16)x[(size_t)n * NODE_IN + k] : (_Float16)0.f;
  }
}

// Fused dual GEMM: out1 = A @ W1 + b1, out2 = A @ W2 + b2 (WMMA, f32 acc).
// Both weight matrices held in LDS in swizzled (fragment-native) layout.
__global__ void __launch_bounds__(128)
k_gemm_node2(const _Float16* __restrict__ A, int kpad, int din,
             const float* __restrict__ W1, const float* __restrict__ b1,
             float* __restrict__ out1,
             const float* __restrict__ W2, const float* __restrict__ b2,
             float* __restrict__ out2) {
  __shared__ __align__(32) _Float16 Wz1[HID * HID];
  __shared__ __align__(32) _Float16 Wz2[HID * HID];
  __shared__ __align__(32) _Float16 Asm[16 * HID];
  int tid = threadIdx.x;
  for (int i = tid; i < kpad * HID; i += blockDim.x) {
    int k = i / HID, n = i % HID;
    int dst = bswz(k, n);
    Wz1[dst] = (k < din) ? (_Float16)W1[i] : (_Float16)0.f;
    Wz2[dst] = (k < din) ? (_Float16)W2[i] : (_Float16)0.f;
  }
  __syncthreads();
  int w = tid >> 5;
  int n0 = w * 32;
  int ntiles = N_NODES / 16;
  int nvec = (16 * kpad) >> 3;  // v8h chunks per tile
  for (int t = blockIdx.x; t < ntiles; t += gridDim.x) {
    int r0 = t * 16;
    const v8h* asrc = (const v8h*)(A + (size_t)r0 * kpad);
    for (int i = tid; i < nvec; i += blockDim.x) ((v8h*)Asm)[i] = asrc[i];
    // prefetch next tile while this one computes
    int tn = t + gridDim.x;
    if (tn < ntiles)
      __builtin_prefetch(A + (size_t)tn * 16 * kpad + tid * 16, 0, 0);
    __syncthreads();
    v8f c0 = {}, c1 = {}, c2 = {}, c3 = {};
    for (int k0 = 0; k0 < kpad; k0 += 32) {
      v16h a = load_a_frag(Asm, kpad, k0);
      c0 = wmma_f16(a, load_b_swz(Wz1, k0, n0), c0);
      c1 = wmma_f16(a, load_b_swz(Wz1, k0, n0 + 16), c1);
      c2 = wmma_f16(a, load_b_swz(Wz2, k0, n0), c2);
      c3 = wmma_f16(a, load_b_swz(Wz2, k0, n0 + 16), c3);
    }
    int l = tid & 31, n = l & 15, hh = l >> 4;
#pragma unroll
    for (int r = 0; r < 8; ++r) {
      int row = r0 + hh * 8 + r;
      out1[(size_t)row * HID + n0 + n]      = c0[r] + b1[n0 + n];
      out1[(size_t)row * HID + n0 + 16 + n] = c1[r] + b1[n0 + 16 + n];
      out2[(size_t)row * HID + n0 + n]      = c2[r] + b2[n0 + n];
      out2[(size_t)row * HID + n0 + 16 + n] = c3[r] + b2[n0 + 16 + n];
    }
    __syncthreads();
  }
}

// Fused: ee = e_tile @ we (WMMA); m = xl[src]+xr[dst]+ee; LeakyReLU; per-head
// dot with att -> logit; atomic max per (dst, head). Wave w == head w.
__global__ void __launch_bounds__(128)
k_logits(const _Float16* __restrict__ ea, const float* __restrict__ we,
         const float* __restrict__ att, const int* __restrict__ ei,
         const float* __restrict__ xl, const float* __restrict__ xr,
         float* __restrict__ logit, float* __restrict__ nodemax) {
  __shared__ __align__(32) _Float16 Wz[HID * HID];
  __shared__ __align__(32) _Float16 Asm[16 * HID];
  __shared__ __align__(16) float xls[16 * HID];
  __shared__ __align__(16) float xrd[16 * HID];
  __shared__ float attsm[NHEAD * CHEAD];
  __shared__ int   ssm[16];
  __shared__ int   dsm[16];
  int tid = threadIdx.x;
  for (int i = tid; i < HID * HID; i += blockDim.x) Wz[bswz(i / HID, i % HID)] = (_Float16)we[i];
  for (int i = tid; i < NHEAD * CHEAD; i += blockDim.x) attsm[i] = att[i];
  __syncthreads();
  int w = tid >> 5;
  int n0 = w * 32;
  int ntiles = EA / 16;
  for (int t = blockIdx.x; t < ntiles; t += gridDim.x) {
    int base = t * 16;
    if (tid < 16) {
      int g = base + tid;
      ssm[tid] = (g < N_EDGES) ? ei[g] : g - N_EDGES;
      dsm[tid] = (g < N_EDGES) ? ei[N_EDGES + g] : g - N_EDGES;
    }
    const v8h* asrc = (const v8h*)(ea + (size_t)base * HID);
    for (int i = tid; i < 256; i += blockDim.x) ((v8h*)Asm)[i] = asrc[i];
    int tn = t + gridDim.x;
    if (tn < ntiles)
      __builtin_prefetch(ea + (size_t)tn * 16 * HID + tid * 16, 0, 0);
    __syncthreads();
    for (int i = tid; i < 16 * (HID / 4); i += blockDim.x) {  // 512 float4
      int r = i >> 5, q = i & 31;
      ((float4*)xls)[i] = ((const float4*)(xl + (size_t)ssm[r] * HID))[q];
      ((float4*)xrd)[i] = ((const float4*)(xr + (size_t)dsm[r] * HID))[q];
    }
    __syncthreads();
    v8f c0 = {}, c1 = {};
#pragma unroll
    for (int k0 = 0; k0 < HID; k0 += 32) {
      v16h a = load_a_frag(Asm, HID, k0);
      c0 = wmma_f16(a, load_b_swz(Wz, k0, n0), c0);
      c1 = wmma_f16(a, load_b_swz(Wz, k0, n0 + 16), c1);
    }
    int l = tid & 31, n = l & 15, hh = l >> 4;
#pragma unroll
    for (int r = 0; r < 8; ++r) {
      int row = hh * 8 + r;
      float m0 = xls[row * HID + n0 + n]      + xrd[row * HID + n0 + n]      + c0[r];
      float m1 = xls[row * HID + n0 + 16 + n] + xrd[row * HID + n0 + 16 + n] + c1[r];
      float s0 = m0 > 0.f ? m0 : SLOPE * m0;
      float s1 = m1 > 0.f ? m1 : SLOPE * m1;
      float p = s0 * attsm[w * CHEAD + n] + s1 * attsm[w * CHEAD + 16 + n];
      p += __shfl_xor(p, 1, 32);
      p += __shfl_xor(p, 2, 32);
      p += __shfl_xor(p, 4, 32);
      p += __shfl_xor(p, 8, 32);
      if (n == 0) {
        logit[(size_t)(base + row) * NHEAD + w] = p;
        atomicMaxF(&nodemax[dsm[row] * NHEAD + w], p);
      }
    }
    __syncthreads();
  }
}

// w = exp(logit - max[dst]); accumulate segment sums (in-place overwrite logit)
__global__ void k_expsum(const int* __restrict__ ei, float* __restrict__ logit,
                         const float* __restrict__ nodemax, float* __restrict__ nsumh) {
  size_t total = (size_t)EA * NHEAD;
  for (size_t i = (size_t)blockIdx.x * blockDim.x + threadIdx.x; i < total;
       i += (size_t)gridDim.x * blockDim.x) {
    int g = (int)(i >> 2), h = (int)(i & 3);
    int d = (g < N_EDGES) ? ei[N_EDGES + g] : g - N_EDGES;
    float wv = __expf(logit[i] - nodemax[d * NHEAD + h]);
    logit[i] = wv;
    atomicAdd(&nsumh[d * NHEAD + h], wv);
  }
}

// acc[dst] += alpha * xl[src]  (xl table is L2-resident); float4 reads
__global__ void k_scatter(const int* __restrict__ ei, const float* __restrict__ wbuf,
                          const float* __restrict__ nsumh, const float* __restrict__ xl,
                          float* __restrict__ acc) {
  size_t total = (size_t)EA * (HID / 4);
  for (size_t i = (size_t)blockIdx.x * blockDim.x + threadIdx.x; i < total;
       i += (size_t)gridDim.x * blockDim.x) {
    int g = (int)(i >> 5), q = (int)(i & 31), h = q >> 3;
    int s = (g < N_EDGES) ? ei[g] : g - N_EDGES;
    int d = (g < N_EDGES) ? ei[N_EDGES + g] : g - N_EDGES;
    float alpha = wbuf[(size_t)g * NHEAD + h] / (nsumh[d * NHEAD + h] + 1e-16f);
    float4 xv = ((const float4*)(xl + (size_t)s * HID))[q];
    float* dst = acc + (size_t)d * HID + q * 4;
    atomicAdd(dst + 0, alpha * xv.x);
    atomicAdd(dst + 1, alpha * xv.y);
    atomicAdd(dst + 2, alpha * xv.z);
    atomicAdd(dst + 3, alpha * xv.w);
  }
}

// hn = silu(layernorm(acc + bias)); h = residual ? h + hn : hn; keep f16 copy
__global__ void __launch_bounds__(128)
k_lnorm(const float* __restrict__ acc, const float* __restrict__ bias,
        const float* __restrict__ gma, const float* __restrict__ bta,
        float* __restrict__ hbuf, _Float16* __restrict__ hhalf, int residual) {
  __shared__ float red[HID];
  int n = blockIdx.x, c = threadIdx.x;
  float v = acc[(size_t)n * HID + c] + bias[c];
  red[c] = v;
  __syncthreads();
  for (int s = 64; s > 0; s >>= 1) { if (c < s) red[c] += red[c + s]; __syncthreads(); }
  float mu = red[0] / (float)HID;
  __syncthreads();
  float dv = v - mu;
  red[c] = dv * dv;
  __syncthreads();
  for (int s = 64; s > 0; s >>= 1) { if (c < s) red[c] += red[c + s]; __syncthreads(); }
  float var = red[0] / (float)HID;
  float y = dv * rsqrtf(var + 1e-5f) * gma[c] + bta[c];
  float sy = siluf(y);
  float hn = residual ? (hbuf[(size_t)n * HID + c] + sy) : sy;
  hbuf[(size_t)n * HID + c] = hn;
  hhalf[(size_t)n * HID + c] = (_Float16)hn;
}

__global__ void k_pool_acc(const float* __restrict__ hbuf, const int* __restrict__ batch,
                           float* __restrict__ psum, float* __restrict__ pmax,
                           float* __restrict__ pcnt) {
  size_t total = (size_t)N_NODES * HID;
  for (size_t i = (size_t)blockIdx.x * blockDim.x + threadIdx.x; i < total;
       i += (size_t)gridDim.x * blockDim.x) {
    int n = (int)(i >> 7), c = (int)(i & 127);
    int g = batch[n];
    float v = hbuf[i];
    atomicAdd(&psum[(size_t)g * HID + c], v);
    atomicMaxF(&pmax[(size_t)g * HID + c], v);
    if (c == 0) atomicAdd(&pcnt[g], 1.f);
  }
}

__global__ void k_pool_fin(const float* __restrict__ psum, const float* __restrict__ pmax,
                           const float* __restrict__ pcnt, float* __restrict__ out) {
  size_t total = (size_t)NGRAPH * HID;
  for (size_t i = (size_t)blockIdx.x * blockDim.x + threadIdx.x; i < total;
       i += (size_t)gridDim.x * blockDim.x) {
    int g = (int)(i >> 7), c = (int)(i & 127);
    out[(size_t)g * 2 * HID + c]       = psum[i] / fmaxf(pcnt[g], 1.f);
    out[(size_t)g * 2 * HID + HID + c] = pmax[i];
  }
}

// ---------------------------------------------------------------------------
extern "C" void kernel_launch(void* const* d_in, const int* in_sizes, int n_in,
                              void* d_out, int out_size, void* d_ws, size_t ws_size,
                              hipStream_t stream) {
  const float* x     = (const float*)d_in[0];
  const int*   ei    = (const int*)d_in[1];
  const float* eattr = (const float*)d_in[2];
  const int*   batch = (const int*)d_in[3];
  const float* epw   = (const float*)d_in[4];
  const float* epb   = (const float*)d_in[5];
  const float *wl[3], *bl[3], *wr[3], *br[3], *wev[3], *attv[3], *biasv[3], *gv[3], *bv[3];
  for (int i = 0; i < 3; ++i) {
    int o = 6 + 9 * i;
    wl[i]   = (const float*)d_in[o + 0]; bl[i] = (const float*)d_in[o + 1];
    wr[i]   = (const float*)d_in[o + 2]; br[i] = (const float*)d_in[o + 3];
    wev[i]  = (const float*)d_in[o + 4]; attv[i] = (const float*)d_in[o + 5];
    biasv[i]= (const float*)d_in[o + 6]; gv[i] = (const float*)d_in[o + 7];
    bv[i]   = (const float*)d_in[o + 8];
  }
  float* out = (float*)d_out;

  char* wp = (char*)d_ws;
  auto take = [&](size_t bytes) -> char* {
    char* p = wp;
    wp += (bytes + 255) & ~(size_t)255;
    return p;
  };
  _Float16* ea    = (_Float16*)take((size_t)EA * HID * 2);
  float*    hbuf  = (float*)take((size_t)N_NODES * HID * 4);
  _Float16* hhalf = (_Float16*)take((size_t)N_NODES * HID * 2);
  _Float16* xpad  = (_Float16*)take((size_t)N_NODES * 32 * 2);
  float* xl    = (float*)take((size_t)N_NODES * HID * 4);
  float* xr    = (float*)take((size_t)N_NODES * HID * 4);
  float* accb  = (float*)take((size_t)N_NODES * HID * 4);
  float* cnt   = (float*)take((size_t)N_NODES * 4);
  float* wbuf  = (float*)take((size_t)EA * NHEAD * 4);
  float* nmax  = (float*)take((size_t)N_NODES * NHEAD * 4);
  float* nsumh = (float*)take((size_t)N_NODES * NHEAD * 4);
  float* psum  = (float*)take((size_t)NGRAPH * HID * 4);
  float* pmax  = (float*)take((size_t)NGRAPH * HID * 4);
  float* pcnt  = (float*)take((size_t)NGRAPH * 4);

  auto fill = [&](float* p, float v, size_t n) {
    k_fill<<<2048, 256, 0, stream>>>(p, v, n);
  };

  // edge projection + scatter-mean self-loop features
  fill(accb, 0.f, (size_t)N_NODES * HID);
  fill(cnt, 0.f, (size_t)N_NODES);
  k_edgeproj<<<N_EDGES / 64, 256, 0, stream>>>(eattr, epw, epb, ei, ea, accb, cnt);
  k_selfloop<<<2048, 256, 0, stream>>>(accb, cnt, ea);
  k_xpad<<<2048, 256, 0, stream>>>(x, xpad);

  for (int L = 0; L < 3; ++L) {
    const _Float16* A = (L == 0) ? xpad : hhalf;
    int kpad = (L == 0) ? 32 : 128;
    int din  = (L == 0) ? NODE_IN : HID;
    k_gemm_node2<<<1024, 128, 0, stream>>>(A, kpad, din, wl[L], bl[L], xl,
                                           wr[L], br[L], xr);
    fill(nmax, -1e30f, (size_t)N_NODES * NHEAD);
    fill(nsumh, 0.f, (size_t)N_NODES * NHEAD);
    fill(accb, 0.f, (size_t)N_NODES * HID);
    k_logits<<<2048, 128, 0, stream>>>(ea, wev[L], attv[L], ei, xl, xr, wbuf, nmax);
    k_expsum<<<2048, 256, 0, stream>>>(ei, wbuf, nmax, nsumh);
    k_scatter<<<4096, 256, 0, stream>>>(ei, wbuf, nsumh, xl, accb);
    k_lnorm<<<N_NODES, 128, 0, stream>>>(accb, biasv[L], gv[L], bv[L], hbuf, hhalf, L > 0);
  }

  fill(psum, 0.f, (size_t)NGRAPH * HID);
  fill(pmax, -1e30f, (size_t)NGRAPH * HID);
  fill(pcnt, 0.f, (size_t)NGRAPH);
  k_pool_acc<<<2048, 256, 0, stream>>>(hbuf, batch, psum, pmax, pcnt);
  k_pool_fin<<<128, 256, 0, stream>>>(psum, pmax, pcnt, out);
}
